// F_ALSHConv2d_7198365188565
// MI455X (gfx1250) — matrix operand order repaired and checked by reading, not compiled, not run on hardware
//
#include <hip/hip_runtime.h>
#include <math.h>

// ---- static problem config (matches reference) ----
#define IN_C   256
#define OUT_C  512
#define KS     3
#define Hdim   56
#define Wdim   56
#define HW     (Hdim*Wdim)          // 3136
#define Bsz    8
#define NH     8
#define TS     1024
#define ROWLEN 8192
#define KTOT   (IN_C*KS*KS)         // 2304
#define AUGROW (KTOT + 18)          // 2322
#define NTOT   (Bsz*HW)             // 25088
#define UFAC   0.83f
#define NKSTEP (KTOT/32)            // 72

typedef __attribute__((ext_vector_type(16))) _Float16 v16h;
typedef __attribute__((ext_vector_type(8)))  float    v8f;

// ---- ws layout ----
// int[0..8191]      counts (NH*TS)
// int[8192]         max sum-of-squares (float bits via int atomicMax)
// int[8193..8200]   buckets
// float[8208..8719] maskf
// byte 36864:       packed f16 weight, k'-ordered (k' = r*256 + c), OUT_C*KTOT halves
#define PACK_OFF   36864
#define PACK_BYTES ((size_t)OUT_C * KTOT * 2)

__global__ void init_kernel(int* p, int n) {
    int i = blockIdx.x * 256 + threadIdx.x;
    if (i < n) p[i] = 0;
}

__global__ void rownorm_kernel(const float* __restrict__ w, unsigned int* denom_bits) {
    __shared__ float red[256];
    const int m = blockIdx.x;
    const float* row = w + (size_t)m * KTOT;
    float s = 0.f;
    for (int i = threadIdx.x; i < KTOT; i += 256) { float v = row[i]; s += v * v; }
    red[threadIdx.x] = s;
    __syncthreads();
    for (int off = 128; off > 0; off >>= 1) {
        if (threadIdx.x < off) red[threadIdx.x] += red[threadIdx.x + off];
        __syncthreads();
    }
    if (threadIdx.x == 0) atomicMax((int*)denom_bits, __float_as_int(red[0]));
}

__global__ void hashvote_kernel(const float* __restrict__ x, const float* __restrict__ ha,
                                const float* __restrict__ hb,
                                const unsigned int* __restrict__ denom_bits,
                                int* __restrict__ counts) {
    int tid = blockIdx.x * 256 + threadIdx.x;
    if (tid >= Bsz * NH * HW) return;
    const int pix  = tid % HW;
    const int rest = tid / HW;
    const int h = rest % NH;
    const int b = rest / NH;
    const int y  = pix / Wdim;
    const int xx = pix % Wdim;
    const float scale = UFAC / sqrtf(__uint_as_float(*denom_bits));
    const float* xb  = x + (size_t)b * IN_C * HW;
    const float* har = ha + (size_t)h * AUGROW;
    float dot = 0.f, augsum = 0.f;
    for (int kh = 0; kh < 3; ++kh) {
        const int iy = y + kh - 1;
        if ((unsigned)iy >= Hdim) continue;
        for (int kw = 0; kw < 3; ++kw) {
            const int ix = xx + kw - 1;
            if ((unsigned)ix >= Wdim) continue;
            const int r = kh * 3 + kw;
            const float* xp = xb + iy * Wdim + ix;
            const float* hp = har + r;
            float acc = 0.f;
            for (int c = 0; c < IN_C; ++c) acc += xp[c * HW] * hp[c * 9];
            dot += acc;
            augsum += har[KTOT + r];   // half-plane (channel 256), zero-padded at borders
        }
    }
    const float dotted = scale * dot + 0.5f * augsum;
    const int idx = ((int)fabsf(floorf(dotted + hb[h]))) % TS;
    atomicAdd(&counts[h * TS + idx], 1);
}

__global__ void bucket_mask_kernel(const int* __restrict__ counts, const int* __restrict__ table,
                                   const int* __restrict__ tlen, float* __restrict__ maskf,
                                   float* __restrict__ outmask, int* __restrict__ buckets) {
    __shared__ int sbuck[NH];
    __shared__ int smask[OUT_C];
    const int t = threadIdx.x;
    if (t < NH) {
        const int* c = counts + t * TS;
        int best = c[0], bi = 0;
        for (int i = 1; i < TS; ++i)
            if (c[i] > best) { best = c[i]; bi = i; }
        sbuck[t] = bi; buckets[t] = bi;
    }
    for (int i = t; i < OUT_C; i += 256) smask[i] = 0;
    __syncthreads();
    for (int h = 0; h < NH; ++h) {
        const int bk = sbuck[h];
        int len = tlen[bk]; if (len > ROWLEN) len = ROWLEN;
        const int* row = table + (size_t)bk * ROWLEN;
        for (int j = t; j < len; j += 256) {
            int e = row[j];
            if ((unsigned)e < OUT_C) smask[e] = 1;
        }
    }
    __syncthreads();
    for (int i = t; i < OUT_C; i += 256) {
        float v = (float)smask[i];
        maskf[i] = v;
        outmask[i] = v;
    }
}

// masked weight -> f16, reordered k' = r*256 + c (tap-major)
__global__ void pack_weight_kernel(const float* __restrict__ w, const float* __restrict__ maskf,
                                   _Float16* __restrict__ wpack) {
    int idx = blockIdx.x * 256 + threadIdx.x;
    if (idx >= OUT_C * KTOT) return;
    const int m = idx / KTOT;
    const int t = idx - m * KTOT;
    const int r = t >> 8;      // tap 0..8
    const int c = t & 255;     // channel
    wpack[idx] = (_Float16)(w[(size_t)m * KTOT + c * 9 + r] * maskf[m]);
}

// ---- main conv as implicit GEMM with v_wmma_f32_16x16x32_f16 ----
// M=512, N=25088, K=2304 (k' tap-major). Block tile 128x128, BK=32, double-buffered
// LDS staged with GLOBAL_LOAD_ASYNC_TO_LDS (ASYNCcnt), overlapped with WMMA.
#define A_STRIDE_U 20   // uints per A row: 16 data (32 f16) + 4 pad -> conflict-free b128 frags
#define B_STRIDE_F 36   // floats per B row: 32 data + 4 pad -> conflict-free b128 frags

union PK  { _Float16 h[2]; unsigned int u; };
union FRG { uint4 q[2]; v16h v; };

template <bool PACKED>
__global__ __launch_bounds__(256) void conv_wmma_kernel(
    const float* __restrict__ x, const float* __restrict__ w,
    const _Float16* __restrict__ wpack,
    const float* __restrict__ bias, const float* __restrict__ maskf,
    float* __restrict__ out) {
    __shared__ unsigned int A_sh[2][128 * A_STRIDE_U];
    __shared__ float        B_sh[2][128 * B_STRIDE_F];

    const int m0  = blockIdx.y * 128;
    const int n0  = blockIdx.x * 128;
    const int tid = threadIdx.x;
    const int lane = tid & 31;
    const int wid  = tid >> 5;
    const int wm = wid & 1;     // 2 waves in M (64 rows each)
    const int wn = wid >> 1;    // 4 waves in N (32 cols each)
    const int sel  = lane >> 4;
    const int lrow = lane & 15;

    v8f acc[4][2];
    #pragma unroll
    for (int i = 0; i < 4; ++i)
        #pragma unroll
        for (int j = 0; j < 2; ++j)
            acc[i][j] = {};

    // A fallback-loader role
    const int am = tid >> 1;
    const int ah = tid & 1;
    const float amask = PACKED ? 0.f : maskf[m0 + am];

    // B-loader role: one pixel, one k-half (16 channels)
    const int nl  = tid & 127;
    const int kh2 = tid >> 7;
    const int n  = n0 + nl;
    const int bb = n / HW;
    const int pp = n % HW;
    const int py = pp / Wdim;
    const int px = pp % Wdim;
    const float* xb = x + (size_t)bb * IN_C * HW;

    auto stageA = [&](int k0, int bsel) {
        if (PACKED) {
            // contiguous 64B/row copy of packed f16 via async b128 DMA
            const unsigned lbase = (unsigned)(uintptr_t)(&A_sh[bsel][0]);
            #pragma unroll
            for (int ch = 0; ch < 2; ++ch) {
                const int chunk = tid * 2 + ch;          // 512 chunks of 16B
                const int row = chunk >> 2;
                const int q   = chunk & 3;
                const unsigned laddr = lbase + row * (A_STRIDE_U * 4) + q * 16;
                const unsigned long long g =
                    (unsigned long long)(uintptr_t)(wpack + (size_t)(m0 + row) * KTOT + k0) + q * 16;
                asm volatile("global_load_async_to_lds_b128 %0, %1, off"
                             :: "v"(laddr), "v"(g) : "memory");
            }
        } else {
            // gather f32 weight at stride 9, convert, mask, ds-store
            const int r  = k0 >> 8;
            const int cb = (k0 & 255) + ah * 16;
            const float* src = w + (size_t)(m0 + am) * KTOT + r;
            unsigned int* dst = &A_sh[bsel][am * A_STRIDE_U + ah * 8];
            #pragma unroll
            for (int q = 0; q < 8; ++q) {
                PK p;
                p.h[0] = (_Float16)(src[(size_t)(cb + 2 * q) * 9] * amask);
                p.h[1] = (_Float16)(src[(size_t)(cb + 2 * q + 1) * 9] * amask);
                dst[q] = p.u;
            }
        }
    };

    auto stageB = [&](int k0, int bsel) {
        const int r  = k0 >> 8;                 // uniform tap for whole step
        const int cb = (k0 & 255) + kh2 * 16;   // 16 consecutive channels
        const int iy = py + (r / 3) - 1;
        const int ix = px + (r % 3) - 1;
        if ((unsigned)iy < (unsigned)Hdim && (unsigned)ix < (unsigned)Wdim) {
            const float* src = xb + (size_t)cb * HW + iy * Wdim + ix;
            const unsigned lbase =
                (unsigned)(uintptr_t)(&B_sh[bsel][nl * B_STRIDE_F + kh2 * 16]);
            #pragma unroll
            for (int q = 0; q < 16; ++q) {
                const unsigned laddr = lbase + q * 4;
                const unsigned long long g =
                    (unsigned long long)(uintptr_t)(src + (size_t)q * HW);
                asm volatile("global_load_async_to_lds_b32 %0, %1, off"
                             :: "v"(laddr), "v"(g) : "memory");
            }
        } else {
            uint4 z = make_uint4(0u, 0u, 0u, 0u);
            uint4* d = reinterpret_cast<uint4*>(&B_sh[bsel][nl * B_STRIDE_F + kh2 * 16]);
            d[0] = z; d[1] = z; d[2] = z; d[3] = z;
        }
    };

    auto compute = [&](int bsel) {
        FRG a[4];
        #pragma unroll
        for (int i = 0; i < 4; ++i) {
            const unsigned int* p = &A_sh[bsel][(wm * 64 + i * 16 + lrow) * A_STRIDE_U];
            a[i].q[0] = *reinterpret_cast<const uint4*>(p + sel * 8);      // K 0..7  | 16..23
            a[i].q[1] = *reinterpret_cast<const uint4*>(p + 4 + sel * 8);  // K 8..15 | 24..31
        }
        #pragma unroll
        for (int j = 0; j < 2; ++j) {
            const float4* bp = reinterpret_cast<const float4*>(
                &B_sh[bsel][(wn * 32 + j * 16 + lrow) * B_STRIDE_F + sel * 16]);
            const float4 f0 = bp[0], f1 = bp[1], f2 = bp[2], f3 = bp[3];
            v16h bv;
            bv[0]  = (_Float16)f0.x; bv[1]  = (_Float16)f0.y;
            bv[2]  = (_Float16)f0.z; bv[3]  = (_Float16)f0.w;
            bv[4]  = (_Float16)f1.x; bv[5]  = (_Float16)f1.y;
            bv[6]  = (_Float16)f1.z; bv[7]  = (_Float16)f1.w;
            bv[8]  = (_Float16)f2.x; bv[9]  = (_Float16)f2.y;
            bv[10] = (_Float16)f2.z; bv[11] = (_Float16)f2.w;
            bv[12] = (_Float16)f3.x; bv[13] = (_Float16)f3.y;
            bv[14] = (_Float16)f3.z; bv[15] = (_Float16)f3.w;
            #pragma unroll
            for (int i = 0; i < 4; ++i)
                acc[i][j] = __builtin_amdgcn_wmma_f32_16x16x32_f16(
                    false, a[i].v, false, bv, (short)0, acc[i][j], false, false);
        }
    };

    // ---- software pipeline: async DMA of tile k+1 overlaps WMMA on tile k ----
    stageA(0, 0);
    stageB(0, 0);
    asm volatile("s_wait_asynccnt 0" ::: "memory");
    __syncthreads();
    for (int kt = 0; kt < NKSTEP; ++kt) {
        const int cur = kt & 1;
        if (kt + 1 < NKSTEP) {
            stageA((kt + 1) * 32, cur ^ 1);
            stageB((kt + 1) * 32, cur ^ 1);
        }
        compute(cur);
        asm volatile("s_wait_asynccnt 0" ::: "memory");
        __syncthreads();
    }

    // ---- store: C layout row M = vgpr + 8*sel, col N = lane&15 ----
    const int ncb = n0 + wn * 32 + lrow;
    #pragma unroll
    for (int j = 0; j < 2; ++j) {
        const int nc = ncb + j * 16;
        const int ob = nc / HW;
        const int op = nc % HW;
        const size_t obase = (size_t)ob * OUT_C * HW + op;
        #pragma unroll
        for (int i = 0; i < 4; ++i) {
            const int mbase = m0 + wm * 64 + i * 16 + sel * 8;
            #pragma unroll
            for (int v = 0; v < 8; ++v) {
                const int mrow = mbase + v;
                out[obase + (size_t)mrow * HW] = (acc[i][j][v] + bias[mrow]) * 128.0f;
            }
        }
    }
}

extern "C" void kernel_launch(void* const* d_in, const int* in_sizes, int n_in,
                              void* d_out, int out_size, void* d_ws, size_t ws_size,
                              hipStream_t stream) {
    (void)in_sizes; (void)n_in; (void)out_size;
    const float* x    = (const float*)d_in[0];
    const float* wgt  = (const float*)d_in[1];
    const float* bias = (const float*)d_in[2];
    const float* ha   = (const float*)d_in[3];
    const float* hb   = (const float*)d_in[4];
    const int*   table = (const int*)d_in[5];
    const int*   tlen  = (const int*)d_in[6];
    float* out = (float*)d_out;

    int*          counts     = (int*)d_ws;
    unsigned int* denom_bits = (unsigned int*)d_ws + 8192;
    int*          buckets    = (int*)d_ws + 8193;
    float*        maskf      = (float*)d_ws + 8208;
    _Float16*     wpack      = (_Float16*)((char*)d_ws + PACK_OFF);
    float*        outmask    = out + (size_t)Bsz * OUT_C * HW;

    // fixed per-harness decision -> deterministic across replays
    const bool packed = ws_size >= PACK_OFF + PACK_BYTES;

    init_kernel<<<(8193 + 255) / 256, 256, 0, stream>>>((int*)d_ws, 8193);
    rownorm_kernel<<<OUT_C, 256, 0, stream>>>(wgt, denom_bits);
    hashvote_kernel<<<(Bsz * NH * HW + 255) / 256, 256, 0, stream>>>(x, ha, hb, denom_bits, counts);
    bucket_mask_kernel<<<1, 256, 0, stream>>>(counts, table, tlen, maskf, outmask, buckets);
    if (packed) {
        pack_weight_kernel<<<(OUT_C * KTOT + 255) / 256, 256, 0, stream>>>(wgt, maskf, wpack);
        conv_wmma_kernel<true><<<dim3(NTOT / 128, OUT_C / 128), 256, 0, stream>>>(
            x, wgt, wpack, bias, maskf, out);
    } else {
        conv_wmma_kernel<false><<<dim3(NTOT / 128, OUT_C / 128), 256, 0, stream>>>(
            x, wgt, wpack, bias, maskf, out);
    }
}
